// MambaBlock_82171314307105
// MI455X (gfx1250) — matrix operand order repaired and checked by reading, not compiled, not run on hardware
//
#include <hip/hip_runtime.h>

typedef __bf16 bf16;
typedef __attribute__((ext_vector_type(16))) __bf16 bf16x16;
typedef __attribute__((ext_vector_type(8)))  __bf16 bf16x8;
typedef __attribute__((ext_vector_type(8)))  float  f32x8;
typedef __attribute__((ext_vector_type(4)))  int    i32x4;

#define D_MODEL 1024
#define SEQ_IN  1022
#define SEQ     1024
#define BATCH   8
#define M_PAD   (BATCH*SEQ)      /* 8192 */
#define M_IN    (BATCH*SEQ_IN)   /* 8176 */

#define BM 128
#define BN 128
#define BK 32
#define AST 40   /* LDS row stride (bf16) for A tile  [BM][AST] */
#define BST 40   /* LDS row stride (bf16) for B^T tile [BN][BST] */

#if defined(__gfx1250__) && __has_builtin(__builtin_amdgcn_global_load_async_to_lds_b128) && __has_builtin(__builtin_amdgcn_s_wait_asynccnt)
#define USE_ASYNC_COPY 1
#else
#define USE_ASYNC_COPY 0
#endif

__device__ __forceinline__ bf16x16 cat8(bf16x8 lo, bf16x8 hi) {
  return __builtin_shufflevector(lo, hi, 0,1,2,3,4,5,6,7,8,9,10,11,12,13,14,15);
}

// ACT: 0 = none, 1 = silu, 2 = mamba-delta: out = exp(-softplus(acc+bias)*avec[col])
// ACC: add accIn[] (f32) to the accumulator before the epilogue.
template<int ACT, bool ACC>
__global__ __launch_bounds__(256) void gemm_bf16_wmma(
    const bf16* __restrict__ A, long sAb,
    const bf16* __restrict__ B, long sBb,
    int M, int N, int K,
    const float* __restrict__ bias,
    const float* __restrict__ avec,
    const float* __restrict__ accIn,
    float* __restrict__ outF,
    bf16* __restrict__ outB,
    long sOb)
{
  __shared__ __align__(16) bf16 Al[2][BM * AST];
  __shared__ __align__(16) bf16 Bl[2][BN * BST];

  const int tid  = threadIdx.x;
  const int lane = tid & 31;
  const int wave = tid >> 5;
  const int wm   = wave & 3;   // wave M index: 0..3  (32 rows each)
  const int wn   = wave >> 2;  // wave N index: 0..1  (64 cols each)

  const long mBlock = (long)blockIdx.y * BM;
  const int  n0     = blockIdx.x * BN;
  const bf16* Ab = A + (long)blockIdx.z * sAb + mBlock * K;
  const bf16* Bb = B + (long)blockIdx.z * sBb;

  // copy geometry:
  //  A tile 128x32 bf16 = 512 x 16B chunks -> 2 chunks / thread (async DMA to LDS)
  //  B tile  32x128 bf16 -> 32B / thread, stored transposed [BN][BST]
  const int brow = tid >> 3, bcol = (tid & 7) * 16;

  f32x8 acc[2][4] = {};

  auto issue = [&](int buf, int k0) {
#pragma unroll
    for (int p = 0; p < 2; ++p) {
      int chunk = tid + 256 * p;
      int arow = chunk >> 2, acol = (chunk & 3) * 8;
      const bf16* gA = Ab + (long)arow * K + (k0 + acol);
#if USE_ASYNC_COPY
      __builtin_amdgcn_global_load_async_to_lds_b128(
          (i32x4*)gA, (i32x4*)&Al[buf][arow * AST + acol], 0, 0);
#else
      bf16x8 va = *(const bf16x8*)gA;
      *(bf16x8*)&Al[buf][arow * AST + acol] = va;
#endif
    }
    const bf16* gB = Bb + (long)(k0 + brow) * N + (n0 + bcol);
    bf16x8 b0 = *(const bf16x8*)gB;
    bf16x8 b1 = *(const bf16x8*)(gB + 8);
#pragma unroll
    for (int i = 0; i < 8; ++i) Bl[buf][(bcol + i)     * BST + brow] = b0[i];
#pragma unroll
    for (int i = 0; i < 8; ++i) Bl[buf][(bcol + 8 + i) * BST + brow] = b1[i];
  };

  const int S = K / BK;
  issue(0, 0);

  const int r   = lane & 15;
  const int hi  = lane >> 4;
  const int kbA = hi ? 8 : 0;    // A frag: lanes 0-15 K{0-7,16-23}, lanes 16-31 K{8-15,24-31}
  const int kbB = hi ? 8 : 0;    // B frag: lanes 0-15 K{0-7,16-23}, lanes 16-31 K{8-15,24-31}

  for (int s = 0; s < S; ++s) {
#if USE_ASYNC_COPY
    __builtin_amdgcn_s_wait_asynccnt(0);
#endif
    __syncthreads();
    if (s + 1 < S) issue((s + 1) & 1, (s + 1) * BK);

    const bf16* Ac = Al[s & 1];
    const bf16* Bc = Bl[s & 1];

    bf16x16 af[2], bfr[4];
#pragma unroll
    for (int i = 0; i < 2; ++i) {
      int row = wm * 32 + i * 16 + r;
      af[i] = cat8(*(const bf16x8*)&Ac[row * AST + kbA],
                   *(const bf16x8*)&Ac[row * AST + kbA + 16]);
    }
#pragma unroll
    for (int j = 0; j < 4; ++j) {
      int col = wn * 64 + j * 16 + r;
      bfr[j] = cat8(*(const bf16x8*)&Bc[col * BST + kbB],
                    *(const bf16x8*)&Bc[col * BST + kbB + 16]);
    }
#pragma unroll
    for (int i = 0; i < 2; ++i)
#pragma unroll
      for (int j = 0; j < 4; ++j)
        acc[i][j] = __builtin_amdgcn_wmma_f32_16x16x32_bf16(
            false, af[i], false, bfr[j], (short)0, acc[i][j], false, false);
  }

  // epilogue (C/D layout: VGPR e -> M = rowb+e, N = col)
  const long obase = (long)blockIdx.z * sOb;
#pragma unroll
  for (int i = 0; i < 2; ++i) {
#pragma unroll
    for (int j = 0; j < 4; ++j) {
      int  col  = n0 + wn * 64 + j * 16 + r;
      long rowb = mBlock + wm * 32 + i * 16 + hi * 8;
      float bval = bias ? bias[col] : 0.f;
      float aval = (ACT == 2) ? avec[col] : 0.f;
#pragma unroll
      for (int e = 0; e < 8; ++e) {
        long off = obase + (rowb + e) * N + col;
        float v = acc[i][j][e];
        if (ACC) v += accIn[off];
        v += bval;
        if (ACT == 1) v = v / (1.f + __expf(-v));
        if (ACT == 2) {
          float sp = (v > 20.f) ? v : log1pf(__expf(v));
          v = __expf(-sp * aval);
        }
        if (outF) outF[off] = v;
        if (outB) outB[off] = (bf16)v;
      }
    }
  }
}

// fp32 -> bf16 convert with zero padding beyond n (pads M 8176 -> 8192)
__global__ void cvt_kernel(const float* __restrict__ src, bf16* __restrict__ dst,
                           int n, int ntot) {
  int i = blockIdx.x * blockDim.x + threadIdx.x;
  if (i >= ntot) return;
  dst[i] = (i < n) ? (bf16)src[i] : (bf16)0.f;
}

// depthwise conv (k=3, pad 2 both sides: T 1022 -> 1024) + bias + SiLU -> bf16
__global__ void conv_silu_kernel(const float* __restrict__ x1,
                                 const float* __restrict__ cw,
                                 const float* __restrict__ cb,
                                 bf16* __restrict__ xs) {
  int idx = blockIdx.x * blockDim.x + threadIdx.x;   // 8*1024*1024
  int d = idx & (D_MODEL - 1);
  int t = (idx >> 10) & (SEQ - 1);
  int b = idx >> 20;
  float acc = cb[d];
#pragma unroll
  for (int k = 0; k < 3; ++k) {
    int ti = t + k - 2;
    if (0 <= ti && ti < SEQ_IN)
      acc += cw[d * 3 + k] * x1[(((long)(b * SEQ_IN + ti)) << 10) + d];
  }
  acc = acc / (1.f + __expf(-acc));
  xs[idx] = (bf16)acc;
}

// selective scan: h_t = abar_t * h_{t-1} + bx_t, per (batch, channel); coalesced in d
__global__ void scan_kernel(const float* __restrict__ abar,
                            const float* __restrict__ bx,
                            bf16* __restrict__ h) {
  int id = blockIdx.x * blockDim.x + threadIdx.x;   // 8*1024
  int d = id & (D_MODEL - 1);
  int b = id >> 10;
  long base = ((long)b << 20) + d;
  float hv = 0.f;
  for (int t = 0; t < SEQ; ++t) {
    long o = base + ((long)t << 10);
    hv = abar[o] * hv + bx[o];
    h[o] = (bf16)hv;
  }
}

static inline size_t align256(size_t x) { return (x + 255) & ~(size_t)255; }

extern "C" void kernel_launch(void* const* d_in, const int* in_sizes, int n_in,
                              void* d_out, int out_size, void* d_ws, size_t ws_size,
                              hipStream_t stream)
{
  (void)in_sizes; (void)n_in; (void)out_size; (void)ws_size;
  const float* x      = (const float*)d_in[0];
  const float* w1     = (const float*)d_in[1];
  const float* w2     = (const float*)d_in[2];
  const float* wlast  = (const float*)d_in[3];
  const float* convw  = (const float*)d_in[4];
  const float* convb  = (const float*)d_in[5];
  const float* Avec   = (const float*)d_in[6];
  const float* wB     = (const float*)d_in[7];
  const float* bB     = (const float*)d_in[8];
  const float* wC     = (const float*)d_in[9];
  const float* bC     = (const float*)d_in[10];
  const float* wD     = (const float*)d_in[11];
  const float* bD     = (const float*)d_in[12];
  const float* wdelta = (const float*)d_in[13];
  const float* bdelta = (const float*)d_in[14];

  char* ws = (char*)d_ws;
  size_t off = 0;
  auto take = [&](size_t bytes) { void* p = ws + off; off = align256(off + bytes); return p; };

  const size_t WMAT = (size_t)D_MODEL * D_MODEL;   // 1M elems
  const size_t MTOK = (size_t)M_PAD * D_MODEL;     // 8M elems
  const long   BSTR = (long)SEQ * D_MODEL;         // per-batch stride

  bf16* w1b = (bf16*)take(WMAT * 2);
  bf16* w2b = (bf16*)take(WMAT * 2);
  bf16* wlb = (bf16*)take(WMAT * 2);
  bf16* wdb = (bf16*)take(WMAT * 2);
  bf16* wBb = (bf16*)take(WMAT * 2);
  bf16* wCb = (bf16*)take(WMAT * 2);
  bf16* wDb = (bf16*)take(WMAT * 2);
  bf16*  xb   = (bf16*)take(MTOK * 2);
  float* x1   = (float*)take(MTOK * 4);
  bf16*  xs   = (bf16*)take(MTOK * 2);
  float* abar = (float*)take(MTOK * 4);
  float* bx   = (float*)take(MTOK * 4);
  bf16*  hb   = (bf16*)take(MTOK * 2);
  bf16*  ssmb = (bf16*)take(MTOK * 2);
  float* ssmf = abar;  // abar fully consumed by the scan before ssm pass 1 writes it
  bf16*  xres = hb;    // hb fully consumed by ssm pass 1 before x_res is produced
  bf16*  zb   = xb;    // xb only used by the first GEMM

  dim3 blk(256);
  // conversions (bf16 feed for WMMA)
  cvt_kernel<<<(int)(MTOK / 256), blk, 0, stream>>>(x, xb, (int)((size_t)M_IN * D_MODEL), (int)MTOK);
  cvt_kernel<<<(int)(WMAT / 256), blk, 0, stream>>>(w1,     w1b, (int)WMAT, (int)WMAT);
  cvt_kernel<<<(int)(WMAT / 256), blk, 0, stream>>>(w2,     w2b, (int)WMAT, (int)WMAT);
  cvt_kernel<<<(int)(WMAT / 256), blk, 0, stream>>>(wlast,  wlb, (int)WMAT, (int)WMAT);
  cvt_kernel<<<(int)(WMAT / 256), blk, 0, stream>>>(wdelta, wdb, (int)WMAT, (int)WMAT);
  cvt_kernel<<<(int)(WMAT / 256), blk, 0, stream>>>(wB,     wBb, (int)WMAT, (int)WMAT);
  cvt_kernel<<<(int)(WMAT / 256), blk, 0, stream>>>(wC,     wCb, (int)WMAT, (int)WMAT);
  cvt_kernel<<<(int)(WMAT / 256), blk, 0, stream>>>(wD,     wDb, (int)WMAT, (int)WMAT);

  dim3 g1(D_MODEL / BN, M_PAD / BM, 1);   // (8, 64, 1)
  // x @ w1 -> x1 (f32, pre-conv)
  gemm_bf16_wmma<0, false><<<g1, blk, 0, stream>>>(xb, 0, w1b, 0, M_PAD, D_MODEL, D_MODEL,
      nullptr, nullptr, nullptr, x1, nullptr, 0);
  // depthwise conv + SiLU -> xs (bf16)
  conv_silu_kernel<<<(int)(MTOK / 256), blk, 0, stream>>>(x1, convw, convb, xs);
  // A_bar = exp(-softplus(xs@wdelta + bdelta) * A) -> abar (f32)
  gemm_bf16_wmma<2, false><<<g1, blk, 0, stream>>>(xs, 0, wdb, 0, M_PAD, D_MODEL, D_MODEL,
      bdelta, Avec, nullptr, abar, nullptr, 0);
  // B_x = xs@wB + bB -> bx (f32)
  gemm_bf16_wmma<0, false><<<g1, blk, 0, stream>>>(xs, 0, wBb, 0, M_PAD, D_MODEL, D_MODEL,
      bB, nullptr, nullptr, bx, nullptr, 0);
  // selective scan -> hb (bf16)
  scan_kernel<<<(BATCH * D_MODEL) / 256, blk, 0, stream>>>(abar, bx, hb);
  // ssm pass 1: h@wC + bC -> ssmf (f32)
  gemm_bf16_wmma<0, false><<<g1, blk, 0, stream>>>(hb, 0, wCb, 0, M_PAD, D_MODEL, D_MODEL,
      bC, nullptr, nullptr, ssmf, nullptr, 0);
  // ssm pass 2: ssmf + xs@wD + bD -> ssmb (bf16)
  gemm_bf16_wmma<0, true><<<g1, blk, 0, stream>>>(xs, 0, wDb, 0, M_PAD, D_MODEL, D_MODEL,
      bD, nullptr, ssmf, nullptr, ssmb, 0);
  // x_res = silu(ssm @ w2) -> xres (bf16)
  gemm_bf16_wmma<1, false><<<g1, blk, 0, stream>>>(ssmb, 0, w2b, 0, M_PAD, D_MODEL, D_MODEL,
      nullptr, nullptr, nullptr, nullptr, xres, 0);
  // z = batched ssm @ x_res -> zb (bf16)
  dim3 gz(D_MODEL / BN, SEQ / BM, BATCH);  // (8, 8, 8)
  gemm_bf16_wmma<0, false><<<gz, blk, 0, stream>>>(ssmb, BSTR, xres, BSTR,
      SEQ, D_MODEL, D_MODEL, nullptr, nullptr, nullptr, nullptr, zb, BSTR);
  // out = z @ w_last -> d_out (f32)
  gemm_bf16_wmma<0, false><<<g1, blk, 0, stream>>>(zb, 0, wlb, 0, M_PAD, D_MODEL, D_MODEL,
      nullptr, nullptr, nullptr, (float*)d_out, nullptr, 0);
}
